// MS_SSA_dend_Conv_20117626814987
// MI455X (gfx1250) — matrix operand order repaired and checked by reading, not compiled, not run on hardware
//
#include <hip/hip_runtime.h>
#include <hip/hip_bf16.h>
#include <stdint.h>

typedef _Float16 h8   __attribute__((ext_vector_type(8)));
typedef _Float16 v16h __attribute__((ext_vector_type(16)));
typedef float    v8f  __attribute__((ext_vector_type(8)));

#define TT 4
#define BB 32
#define CC 384
#define NN 256
#define KSTEPS (CC / 32)
// elements of first output (T*B*C*H*W)
#define OUT_ELEMS (TT * BB * CC * NN)

__device__ __forceinline__ v16h cat16(h8 lo, h8 hi) {
  v16h r;
#pragma unroll
  for (int i = 0; i < 8; ++i) { r[i] = lo[i]; r[i + 8] = hi[i]; }
  return r;
}

// A fragment: 16 consecutive f16 (K-half selected via pointer)
__device__ __forceinline__ v16h loadA(const _Float16* p) {
  return cat16(*(const h8*)(p), *(const h8*)(p + 8));
}

// B fragment: 16 consecutive f16 (K-half selected by caller via pointer)
__device__ __forceinline__ v16h loadB(const _Float16* p) {
  return cat16(*(const h8*)(p), *(const h8*)(p + 8));
}

// ---------------------------------------------------------------------------
// Convert f32 weight matrix to f16 (row-major [o][c])
// ---------------------------------------------------------------------------
__global__ void f32_to_f16_kernel(const float* __restrict__ src,
                                  _Float16* __restrict__ dst, int n) {
  int i = blockIdx.x * blockDim.x + threadIdx.x;
  if (i < n) dst[i] = (_Float16)src[i];
}

// ---------------------------------------------------------------------------
// Precompute BN scale/shift for q,k,v and proj (proj folds bias in).
// Layout in out: [q_sc, q_sh, k_sc, k_sh, v_sc, v_sh, p_sc, p_sh] x CC floats
// ---------------------------------------------------------------------------
__global__ void prep_bn_kernel(const float* qg, const float* qb, const float* qm, const float* qv,
                               const float* kg, const float* kb, const float* km, const float* kvv,
                               const float* vg, const float* vb, const float* vm, const float* vv,
                               const float* pg, const float* pb, const float* pm, const float* pv,
                               const float* pbias, float* __restrict__ outp) {
  int c = blockIdx.x * blockDim.x + threadIdx.x;
  if (c >= CC) return;
  const float eps = 1e-5f;
  float s;
  s = qg[c] * rsqrtf(qv[c] + eps);  outp[0 * CC + c] = s; outp[1 * CC + c] = qb[c] - qm[c] * s;
  s = kg[c] * rsqrtf(kvv[c] + eps); outp[2 * CC + c] = s; outp[3 * CC + c] = kb[c] - km[c] * s;
  s = vg[c] * rsqrtf(vv[c] + eps);  outp[4 * CC + c] = s; outp[5 * CC + c] = vb[c] - vm[c] * s;
  s = pg[c] * rsqrtf(pv[c] + eps);  outp[6 * CC + c] = s;
  outp[7 * CC + c] = pbias[c] * s + (pb[c] - pm[c] * s);
}

// ---------------------------------------------------------------------------
// Shortcut LIF on x, with LDS transpose: x [T,B,C,N] f32 -> spikes [T,B,N,C] f16
// One block = 16(c) x 16(n) tile of one image; membrane held per-thread over T.
// ---------------------------------------------------------------------------
__global__ void lif_shortcut_kernel(const float* __restrict__ x,
                                    _Float16* __restrict__ xs_t) {
  __shared__ _Float16 tile[16][18];
  const int n0 = blockIdx.x * 16;
  const int c0 = blockIdx.y * 16;
  const int b  = blockIdx.z;
  const int tid = threadIdx.x;
  const int nx = tid & 15, cy = tid >> 4;   // compute mapping (coalesced x read)
  const int cx = tid & 15, ny = tid >> 4;   // store mapping (coalesced f16 write)
  float vm = 0.f;
  for (int t = 0; t < TT; ++t) {
    size_t xi = ((size_t)(t * BB + b) * CC + (c0 + cy)) * NN + (n0 + nx);
    float xv = x[xi];
    vm = (vm + xv) * 0.5f;                  // v += (x - v)/tau, tau = 2
    float s = (vm >= 1.f) ? 1.f : 0.f;
    vm = (s > 0.f) ? 0.f : vm;              // hard reset
    tile[cy][nx] = (_Float16)s;
    __syncthreads();
    size_t oi = ((size_t)(t * BB + b) * NN + (n0 + ny)) * CC + (c0 + cx);
    xs_t[oi] = tile[cx][ny];
    __syncthreads();
  }
}

// ---------------------------------------------------------------------------
// Fused branch: 1x1 conv (WMMA f16->f32) + BN + LIF over T.
// k-outer / t-inner with explicit one-stage software pipeline: while the 4
// WMMAs of step ks execute, the A and 4 B fragments of ks+1 are in flight,
// so waits are partial instead of loadcnt==0. Register use bounded
// (32 acc + 2*8 A + 2*32 B ~ 110 VGPRs) -> no spills, weights loaded once.
// LIF recurrence runs in the epilogue on the 4 in-register conv results.
// Grid: (n_tile=16, o_group=3, b=32), block = 256 (8 waves, one o-tile each).
// ---------------------------------------------------------------------------
__global__ void branch_gemm_lif_kernel(const _Float16* __restrict__ xs_t,
                                       const _Float16* __restrict__ Wh,
                                       const float* __restrict__ bscale,
                                       const float* __restrict__ bshift,
                                       _Float16* __restrict__ sp_t,
                                       float* __restrict__ vout, int write_v) {
  const int b    = blockIdx.z;
  const int n0   = blockIdx.x * 16;
  const int wave = threadIdx.x >> 5;
  const int o0   = (blockIdx.y * 8 + wave) * 16;
  const int lane = threadIdx.x & 31;
  const int lm   = lane & 15;     // A: row M / B: col N / D: col N
  const int lh   = lane >> 4;     // high half selector

  const _Float16* wrow = Wh + (size_t)(o0 + lm) * CC + lh * 16;
  const size_t tstride = (size_t)BB * NN * CC;
  const _Float16* brow = xs_t + ((size_t)b * NN + (n0 + lm)) * CC + lh * 16;

  v8f acc[TT];
#pragma unroll
  for (int t = 0; t < TT; ++t) acc[t] = (v8f){0.f,0.f,0.f,0.f,0.f,0.f,0.f,0.f};

  // pipeline prologue: fragments for ks = 0
  v16h a = loadA(wrow);
  v16h bcur[TT];
#pragma unroll
  for (int t = 0; t < TT; ++t) bcur[t] = loadB(brow + t * tstride);

#pragma unroll
  for (int ks = 0; ks < KSTEPS; ++ks) {
    v16h an = a;
    v16h bn[TT];
#pragma unroll
    for (int t = 0; t < TT; ++t) bn[t] = bcur[t];
    if (ks + 1 < KSTEPS) {                   // issue next-step loads first
      const int kn = (ks + 1) * 32;
      an = loadA(wrow + kn);
#pragma unroll
      for (int t = 0; t < TT; ++t) bn[t] = loadB(brow + t * tstride + kn);
    }
#pragma unroll
    for (int t = 0; t < TT; ++t)
      acc[t] = __builtin_amdgcn_wmma_f32_16x16x32_f16(
          false, a, false, bcur[t], (short)0, acc[t], false, false);
    a = an;
#pragma unroll
    for (int t = 0; t < TT; ++t) bcur[t] = bn[t];
  }

  float sc[8], sh[8];
#pragma unroll
  for (int r = 0; r < 8; ++r) {
    int o = o0 + r + lh * 8;
    sc[r] = bscale[o];
    sh[r] = bshift[o];
  }

  // Sequential LIF over the 4 in-register conv results
  h8    spv[TT];
  float fs[TT][8];
#pragma unroll
  for (int r = 0; r < 8; ++r) {
    float vm = 0.f;
#pragma unroll
    for (int t = 0; t < TT; ++t) {
      float pre = acc[t][r] * sc[r] + sh[r];  // BN
      vm = (vm + pre) * 0.5f;                 // LIF charge
      float s = (vm >= 1.f) ? 1.f : 0.f;
      vm = (s > 0.f) ? 0.f : vm;
      fs[t][r]  = s;
      spv[t][r] = (_Float16)s;
    }
  }

#pragma unroll
  for (int t = 0; t < TT; ++t) {
    size_t ob = ((size_t)(t * BB + b) * NN + (n0 + lm)) * CC + o0 + lh * 8;
    *(h8*)(sp_t + ob) = spv[t];               // 16B store, 8 consecutive c
  }
  if (write_v) {                              // second output: v in [T,B,8,N,48] f32
    int n = n0 + lm;
#pragma unroll
    for (int t = 0; t < TT; ++t) {
#pragma unroll
      for (int r = 0; r < 8; ++r) {
        int c = o0 + r + lh * 8;
        int h = c / 48, d = c % 48;
        size_t vi = (((size_t)(t * BB + b) * 8 + h) * NN + n) * 48 + d;
        vout[vi] = fs[t][r];
      }
    }
  }
}

// ---------------------------------------------------------------------------
// kv[t,b,c] = sum_n k_spike*v_spike, then LIF (v_th = 0.5) over T.
// One block per b, thread per channel c; coalesced f16 reads across c.
// ---------------------------------------------------------------------------
__global__ void kv_reduce_lif_kernel(const _Float16* __restrict__ ks_t,
                                     const _Float16* __restrict__ vs_t,
                                     _Float16* __restrict__ kvs) {
  const int b = blockIdx.x;
  const int c = threadIdx.x;
  float vm = 0.f;
  for (int t = 0; t < TT; ++t) {
    size_t base = (size_t)(t * BB + b) * NN * CC + c;
    float s = 0.f;
    for (int n = 0; n < NN; ++n)
      s += (float)ks_t[base + (size_t)n * CC] * (float)vs_t[base + (size_t)n * CC];
    vm = (vm + s) * 0.5f;
    float sp = (vm >= 0.5f) ? 1.f : 0.f;
    vm = (sp > 0.f) ? 0.f : vm;
    kvs[(t * BB + b) * CC + c] = (_Float16)sp;
  }
}

// ---------------------------------------------------------------------------
// Proj: y = q_spike * kv_spike (built in-register as B fragment), WMMA GEMM,
// then bias+BN (folded) + residual, f32 stores to d_out. Same one-stage
// software pipeline as the branch kernel.
// Grid: (n_tile=16, o_group=3, tb=128), block = 256.
// ---------------------------------------------------------------------------
__global__ void proj_gemm_out_kernel(const _Float16* __restrict__ qs_t,
                                     const _Float16* __restrict__ kvs,
                                     const _Float16* __restrict__ Whp,
                                     const float* __restrict__ pscale,
                                     const float* __restrict__ pshift,
                                     const float* __restrict__ x,
                                     float* __restrict__ out) {
  const int tb   = blockIdx.z;
  const int n0   = blockIdx.x * 16;
  const int wave = threadIdx.x >> 5;
  const int o0   = (blockIdx.y * 8 + wave) * 16;
  const int lane = threadIdx.x & 31;
  const int lm   = lane & 15;
  const int lh   = lane >> 4;

  const _Float16* wrow = Whp + (size_t)(o0 + lm) * CC + lh * 16;
  const _Float16* qrow = qs_t + ((size_t)tb * NN + (n0 + lm)) * CC + lh * 16;
  const _Float16* grow = kvs + (size_t)tb * CC + lh * 16;

  v8f acc = {0.f, 0.f, 0.f, 0.f, 0.f, 0.f, 0.f, 0.f};
  v16h a = loadA(wrow);
  v16h q = loadB(qrow);
  v16h g = loadB(grow);
#pragma unroll
  for (int ks = 0; ks < KSTEPS; ++ks) {
    v16h an = a, qn = q, gn = g;
    if (ks + 1 < KSTEPS) {                   // next-step loads in flight
      const int kn = (ks + 1) * 32;
      an = loadA(wrow + kn);
      qn = loadB(qrow + kn);
      gn = loadB(grow + kn);
    }
    v16h y = q * g;                          // spike AND spike -> exact f16
    acc = __builtin_amdgcn_wmma_f32_16x16x32_f16(
        false, a, false, y, (short)0, acc, false, false);
    a = an; q = qn; g = gn;
  }
#pragma unroll
  for (int r = 0; r < 8; ++r) {
    int o = o0 + r + lh * 8;
    size_t idx = ((size_t)tb * CC + o) * NN + (n0 + lm);
    out[idx] = acc[r] * pscale[o] + pshift[o] + x[idx];  // bias+BN folded, +identity
  }
}

// ---------------------------------------------------------------------------
extern "C" void kernel_launch(void* const* d_in, const int* in_sizes, int n_in,
                              void* d_out, int out_size, void* d_ws, size_t ws_size,
                              hipStream_t stream) {
  const float* x      = (const float*)d_in[0];
  const float* q_w    = (const float*)d_in[1];
  const float* k_w    = (const float*)d_in[6];
  const float* v_w    = (const float*)d_in[11];
  const float* p_w    = (const float*)d_in[16];
  const float* p_bias = (const float*)d_in[17];

  char* ws = (char*)d_ws;
  _Float16* WhQ = (_Float16*)(ws + 0);
  _Float16* WhK = (_Float16*)(ws + 294912);
  _Float16* WhV = (_Float16*)(ws + 589824);
  _Float16* WhP = (_Float16*)(ws + 884736);
  float*    bnb = (float*)   (ws + 1179648);   // 8*CC floats
  _Float16* XS  = (_Float16*)(ws + 1191936);   // [T,B,N,C] f16
  _Float16* QS  = (_Float16*)(ws + 26357760);
  _Float16* KS  = (_Float16*)(ws + 51523584);
  _Float16* VS  = (_Float16*)(ws + 76689408);
  _Float16* KVS = (_Float16*)(ws + 101855232); // [T,B,C] f16

  float* out  = (float*)d_out;
  float* vout = out + OUT_ELEMS;

  const int wn = CC * CC;
  f32_to_f16_kernel<<<(wn + 255) / 256, 256, 0, stream>>>(q_w, WhQ, wn);
  f32_to_f16_kernel<<<(wn + 255) / 256, 256, 0, stream>>>(k_w, WhK, wn);
  f32_to_f16_kernel<<<(wn + 255) / 256, 256, 0, stream>>>(v_w, WhV, wn);
  f32_to_f16_kernel<<<(wn + 255) / 256, 256, 0, stream>>>(p_w, WhP, wn);

  prep_bn_kernel<<<(CC + 127) / 128, 128, 0, stream>>>(
      (const float*)d_in[2],  (const float*)d_in[3],  (const float*)d_in[4],  (const float*)d_in[5],
      (const float*)d_in[7],  (const float*)d_in[8],  (const float*)d_in[9],  (const float*)d_in[10],
      (const float*)d_in[12], (const float*)d_in[13], (const float*)d_in[14], (const float*)d_in[15],
      (const float*)d_in[18], (const float*)d_in[19], (const float*)d_in[20], (const float*)d_in[21],
      p_bias, bnb);

  lif_shortcut_kernel<<<dim3(NN / 16, CC / 16, BB), 256, 0, stream>>>(x, XS);

  branch_gemm_lif_kernel<<<dim3(16, 3, BB), 256, 0, stream>>>(
      XS, WhQ, bnb + 0 * CC, bnb + 1 * CC, QS, nullptr, 0);
  branch_gemm_lif_kernel<<<dim3(16, 3, BB), 256, 0, stream>>>(
      XS, WhK, bnb + 2 * CC, bnb + 3 * CC, KS, nullptr, 0);
  branch_gemm_lif_kernel<<<dim3(16, 3, BB), 256, 0, stream>>>(
      XS, WhV, bnb + 4 * CC, bnb + 5 * CC, VS, vout, 1);

  kv_reduce_lif_kernel<<<BB, CC, 0, stream>>>(KS, VS, KVS);

  proj_gemm_out_kernel<<<dim3(16, 3, TT * BB), 256, 0, stream>>>(
      QS, KVS, WhP, bnb + 6 * CC, bnb + 7 * CC, x, out);
}
